// SE3Refine_3083786519075
// MI455X (gfx1250) — matrix operand order repaired and checked by reading, not compiled, not run on hardware
//
#include <hip/hip_runtime.h>
#include <math.h>

// ---------------- problem constants (fixed by setup_inputs) ----------------
constexpr int kNR = 1200;
constexpr int kNL = 300;
constexpr int kN  = kNR + kNL;        // 1500
constexpr int kC  = 32;               // rec/lig channels
constexpr int kEMB = 64;              // cross channels
constexpr int kER = kNR * 16;         // 19200
constexpr int kEL = kNL * 16;         // 4800
constexpr int kEC = kNR * kNL;        // 360000
constexpr int kECROSS = kER + kEL + 2 * kEC;  // 744000
constexpr float kEPS = 1e-8f;

typedef __attribute__((ext_vector_type(16))) _Float16 v16h;
typedef __attribute__((ext_vector_type(2)))  __fp16   v2fp16;  // cvt_pkrtz result type
typedef __attribute__((ext_vector_type(8)))  float    v8f;

static inline int ceil_div(int a, int b) { return (a + b - 1) / b; }

// ---------------- small utility kernels ----------------
__global__ void k_fill(float* p, float v, int n) {
  int i = blockIdx.x * blockDim.x + threadIdx.x;
  if (i < n) p[i] = v;
}
__global__ void k_copy(float* dst, const float* src, int n) {
  int i = blockIdx.x * blockDim.x + threadIdx.x;
  if (i < n) dst[i] = src[i];
}

// float atomic max via int/uint punning (works for mixed signs with -3e38 init)
__device__ __forceinline__ void atomicMaxF(float* addr, float v) {
  if (v >= 0.f) atomicMax((int*)addr, __float_as_int(v));
  else          atomicMin((unsigned int*)addr, __float_as_uint(v));
}

// ---- CDNA5 async global->LDS staging (ASYNCcnt-tracked DMA-style copy) ----
__device__ __forceinline__ void async_b32_to_lds(unsigned lds_off, const float* g) {
  asm volatile("global_load_async_to_lds_b32 %0, %1, off"
               :: "v"(lds_off), "v"(g) : "memory");
}
__device__ __forceinline__ void wait_asynccnt0() {
  asm volatile("s_wait_asynccnt 0x0" ::: "memory");
}

// ---------------- WMMA GEMM: C[MxN] = A[MxK] @ B[KxN] (+res), f32 io, f16 compute ----
// One wave computes one 16x16 output tile; 4 waves per block; K in {32,64} (templated,
// fully unrolled so K=64 emits two chained v_wmma). Tiles staged to LDS as f32 via
// global_load_async_to_lds_b32 (B transposed), fragments built with v_cvt_pk_rtz_f16_f32.
template <int K>
__global__ void k_gemm_wmma(const float* __restrict__ A, const float* __restrict__ B,
                            const float* __restrict__ res, float* __restrict__ C,
                            int M, int Nn) {
  __shared__ float sA[4][16 * 32];   // [m][k] tile, f32
  __shared__ float sB[4][16 * 32];   // transposed [n][k] tile, f32
  const int wave = threadIdx.x >> 5;
  const int lane = threadIdx.x & 31;
  const int tilesN = Nn >> 4;
  const int total  = ((M + 15) >> 4) * tilesN;
  int tile = blockIdx.x * 4 + wave;
  const bool active = tile < total;
  if (!active) tile = 0;                      // keep all waves in lockstep (barriers)
  const int tm = (tile / tilesN) << 4;
  const int tn = (tile % tilesN) << 4;
  float* sa = sA[wave];
  float* sb = sB[wave];
  const int hi = lane >> 4;                   // 0/1: lane half
  const int lo = lane & 15;
  const int ar = lane >> 1;                   // A staging: row 0..15
  const int ah = (lane & 1) * 16;             //            col half
  v8f acc = {0.f, 0.f, 0.f, 0.f, 0.f, 0.f, 0.f, 0.f};

#pragma unroll
  for (int k0 = 0; k0 < K; k0 += 32) {
    // ---- stage A[16x32] (async; zero-fill OOB rows via LDS store) ----
    {
      const int gr = tm + ar;
      const bool inb = gr < M;
      const float* ga = A + gr * K + k0 + ah;
      unsigned la = (unsigned)(unsigned long long)(sa + ar * 32 + ah);
#pragma unroll
      for (int j = 0; j < 16; ++j) {
        if (inb) async_b32_to_lds(la + 4u * j, ga + j);
        else     sa[ar * 32 + ah + j] = 0.f;
      }
    }
    // ---- stage B[32x16] transposed into [n][k] (async scatter) ----
    {
      const float* gb = B + (k0 + hi * 16) * Nn + tn + lo;
      unsigned lb = (unsigned)(unsigned long long)(sb + lo * 32 + hi * 16);
#pragma unroll
      for (int j = 0; j < 16; ++j)
        async_b32_to_lds(lb + 4u * j, gb + j * Nn);
    }
    wait_asynccnt0();
    __syncthreads();

    // ---- build f16 fragments with packed converts (ISA 7.12.2 layouts) ----
    v16h af, bf;
    v2fp16* afp = (v2fp16*)&af;
    v2fp16* bfp = (v2fp16*)&bf;
    const float* sarow = sa + lo * 32;        // A row m=lo
    const float* sbrow = sb + lo * 32;        // B col n=lo (transposed row)
#pragma unroll
    for (int t = 0; t < 4; ++t) {
      // af[j]   : K = hi*8 + j      ; af[8+j]: K = 16 + hi*8 + j
      afp[t]     = __builtin_amdgcn_cvt_pkrtz(sarow[hi * 8 + 2 * t],
                                              sarow[hi * 8 + 2 * t + 1]);
      afp[4 + t] = __builtin_amdgcn_cvt_pkrtz(sarow[16 + hi * 8 + 2 * t],
                                              sarow[16 + hi * 8 + 2 * t + 1]);
      // bf[j]   : K = hi*8 + j      ; bf[8+j]: K = 16 + hi*8 + j, N = lo
      bfp[t]     = __builtin_amdgcn_cvt_pkrtz(sbrow[hi * 8 + 2 * t],
                                              sbrow[hi * 8 + 2 * t + 1]);
      bfp[4 + t] = __builtin_amdgcn_cvt_pkrtz(sbrow[16 + hi * 8 + 2 * t],
                                              sbrow[16 + hi * 8 + 2 * t + 1]);
    }
    acc = __builtin_amdgcn_wmma_f32_16x16x32_f16(false, af, false, bf,
                                                 (short)0, acc, false, false);
    __syncthreads();
  }

  if (active) {
#pragma unroll
    for (int r = 0; r < 8; ++r) {             // C/D layout: m = r + 8*hi, n = lo
      int gm = tm + r + hi * 8;
      if (gm < M) {
        int off = gm * Nn + tn + lo;
        float v = acc[r];
        if (res) v += res[off];
        C[off] = v;
      }
    }
  }
}

static inline void gemm(const float* A, const float* B, const float* res, float* C,
                        int M, int Nn, int K, hipStream_t st) {
  int tiles = ((M + 15) / 16) * (Nn / 16);
  int blocks = ceil_div(tiles, 4);
  if (K == 32) k_gemm_wmma<32><<<blocks, 128, 0, st>>>(A, B, res, C, M, Nn);
  else         k_gemm_wmma<64><<<blocks, 128, 0, st>>>(A, B, res, C, M, Nn);
}

// ---------------- edge geometry ----------------
__global__ void k_geom(const float* x, const float* ew, const int* src, const int* dst,
                       const float* side, float* efeat, float* vecdot, int E) {
  int e = blockIdx.x * blockDim.x + threadIdx.x;
  if (e >= E) return;
  int s = src[e], d = dst[e];
  float dx = x[d * 3 + 0] - x[s * 3 + 0];
  float dy = x[d * 3 + 1] - x[s * 3 + 1];
  float dz = x[d * 3 + 2] - x[s * 3 + 2];
  float dist = sqrtf(dx * dx + dy * dy + dz * dz + 1e-12f);
  efeat[e * 5 + 0] = ew[e * 4 + 0];
  efeat[e * 5 + 1] = ew[e * 4 + 1];
  efeat[e * 5 + 2] = ew[e * 4 + 2];
  efeat[e * 5 + 3] = ew[e * 4 + 3];
  efeat[e * 5 + 4] = dist;
  if (side && vecdot) {
    float inv = 1.f / (dist + kEPS);
    vecdot[e] = (side[s * 3 + 0] * dx + side[s * 3 + 1] * dy + side[s * 3 + 2] * dz) * inv;
  }
}

// implicit cross-graph edge decode: [rec edges | lig edges | rec->lig dense | lig->rec dense]
__device__ __forceinline__ void cross_edge_idx(int e, const int* rs, const int* rd,
                                               const int* ls, const int* ld,
                                               int& s, int& d, int& et) {
  if (e < kER)              { s = rs[e];                 d = rd[e];                 et = 0; }
  else if (e < kER + kEL)   { int t = e - kER;           s = ls[t] + kNR; d = ld[t] + kNR; et = 1; }
  else if (e < kER + kEL + kEC) { int t = e - kER - kEL; s = t / kNL;     d = t % kNL + kNR; et = 2; }
  else                      { int t = e - kER - kEL - kEC; s = t % kNL + kNR; d = t / kNL;  et = 2; }
}

__global__ void k_cross_geom(const float* x, const int* rs, const int* rd,
                             const int* ls, const int* ld, float* distC) {
  int e = blockIdx.x * blockDim.x + threadIdx.x;
  if (e >= kECROSS) return;
  int s, d, et; cross_edge_idx(e, rs, rd, ls, ld, s, d, et);
  float dx = x[d * 3 + 0] - x[s * 3 + 0];
  float dy = x[d * 3 + 1] - x[s * 3 + 1];
  float dz = x[d * 3 + 2] - x[s * 3 + 2];
  distC[e] = sqrtf(dx * dx + dy * dy + dz * dz + 1e-12f);
}

// ---------------- fused attention edge passes ----------------
// pass1: fused radial MLP -> rk, logit = dot(q[dst], k[src]*rk)*scale, atomic seg-max
template <bool CROSS>
__global__ void k_attn_pass1(const float* efd, const int* s1, const int* d1,
                             const int* s2, const int* d2,
                             const float* q, const float* kn,
                             const float* w1, const float* b1,
                             const float* w2, const float* b2,
                             float* logit, float* mbuf, int E, int C, float scale) {
  int e = blockIdx.x * blockDim.x + threadIdx.x;
  if (e >= E) return;
  constexpr int EDIM = CROSS ? 4 : 5;
  int s, d; float ef[5];
  if (!CROSS) {
    s = s1[e]; d = d1[e];
#pragma unroll
    for (int t = 0; t < 5; ++t) ef[t] = efd[e * 5 + t];
  } else {
    int et; cross_edge_idx(e, s1, d1, s2, d2, s, d, et);
    ef[0] = (et == 0) ? 1.f : 0.f; ef[1] = (et == 1) ? 1.f : 0.f;
    ef[2] = (et == 2) ? 1.f : 0.f; ef[3] = efd[e];
  }
  float hid[32];
  for (int j = 0; j < 32; ++j) {
    float a = b1[j];
    for (int t = 0; t < EDIM; ++t) a += ef[t] * w1[t * 32 + j];
    hid[j] = a > 0.f ? a : 0.f;
  }
  const float* qd = q + d * C;
  const float* ks = kn + s * C;
  float acc = 0.f;
  for (int c = 0; c < C; ++c) {
    float rk = b2[c];
    for (int j = 0; j < 32; ++j) rk += hid[j] * w2[j * (2 * C) + c];
    acc += qd[c] * ks[c] * rk;
  }
  float lg = acc * scale;
  logit[e] = lg;
  atomicMaxF(mbuf + d, lg);
}

// pass2: e = exp(logit - m[dst]); atomic seg-sum
template <bool CROSS>
__global__ void k_attn_pass2(const int* s1, const int* d1, const int* s2, const int* d2,
                             float* logit, const float* mbuf, float* sbuf, int E) {
  int e = blockIdx.x * blockDim.x + threadIdx.x;
  if (e >= E) return;
  int d;
  if (!CROSS) d = d1[e];
  else { int s, et; cross_edge_idx(e, s1, d1, s2, d2, s, d, et); }
  float ev = __expf(logit[e] - mbuf[d]);
  logit[e] = ev;
  atomicAdd(sbuf + d, ev);
}

// pass3: recompute radial -> rv, a = e/(s+eps), scatter a*v (optionally + vecdot path)
template <bool CROSS>
__global__ void k_attn_pass3(const float* efd, const int* s1, const int* d1,
                             const int* s2, const int* d2,
                             const float* vn,
                             const float* w1, const float* b1,
                             const float* w2, const float* b2,
                             const float* vecdot, const float* rvw1, const float* rvb1,
                             const float* rvw2, const float* rvb2,
                             const float* logit, const float* sbuf,
                             float* msg, int E, int C) {
  int e = blockIdx.x * blockDim.x + threadIdx.x;
  if (e >= E) return;
  constexpr int EDIM = CROSS ? 4 : 5;
  int s, d; float ef[5];
  if (!CROSS) {
    s = s1[e]; d = d1[e];
#pragma unroll
    for (int t = 0; t < 5; ++t) ef[t] = efd[e * 5 + t];
  } else {
    int et; cross_edge_idx(e, s1, d1, s2, d2, s, d, et);
    ef[0] = (et == 0) ? 1.f : 0.f; ef[1] = (et == 1) ? 1.f : 0.f;
    ef[2] = (et == 2) ? 1.f : 0.f; ef[3] = efd[e];
  }
  float hid[32];
  for (int j = 0; j < 32; ++j) {
    float a = b1[j];
    for (int t = 0; t < EDIM; ++t) a += ef[t] * w1[t * 32 + j];
    hid[j] = a > 0.f ? a : 0.f;
  }
  float vd = 0.f;
  float hid2[32];
  if (vecdot) {
    vd = vecdot[e];
    for (int j = 0; j < 32; ++j) {
      float a = rvb1[j];
      for (int t = 0; t < EDIM; ++t) a += ef[t] * rvw1[t * 32 + j];
      hid2[j] = a > 0.f ? a : 0.f;
    }
  }
  float a = logit[e] / (sbuf[d] + kEPS);
  const float* vs = vn + s * C;
  for (int c = 0; c < C; ++c) {
    float rv = b2[C + c];
    for (int j = 0; j < 32; ++j) rv += hid[j] * w2[j * (2 * C) + C + c];
    float vv = vs[c] * rv;
    if (vecdot) {
      float rr = rvb2[c];
      for (int j = 0; j < 32; ++j) rr += hid2[j] * rvw2[j * C + c];
      vv += vd * rr;
    }
    atomicAdd(msg + d * C + c, a * vv);
  }
}

// ---------------- GNormSE3 on degree-0 ----------------
__global__ void k_gnorm(const float* hin, const float* g, const float* b,
                        float* hout, int n, int C) {
  int i = blockIdx.x * blockDim.x + threadIdx.x;
  if (i >= n) return;
  int c = i % C;
  float h = hin[i];
  float nrm = fabsf(h);
  float ph = h / (nrm + kEPS);
  float t = nrm * g[c] + b[c];
  hout[i] = (t > 0.f ? t : 0.f) * ph;
}

// ---------------- GConvSE3 message (scalar out) ----------------
__global__ void k_conv_msg(const float* efeat, const int* src, const int* dst,
                           const float* h, const float* w1, const float* b1,
                           const float* w2, const float* b2, float* msg, int E, int Cin) {
  int e = blockIdx.x * blockDim.x + threadIdx.x;
  if (e >= E) return;
  int s = src[e], d = dst[e];
  float ef[5];
#pragma unroll
  for (int t = 0; t < 5; ++t) ef[t] = efeat[e * 5 + t];
  float hid[32];
  for (int j = 0; j < 32; ++j) {
    float a = b1[j];
    for (int t = 0; t < 5; ++t) a += ef[t] * w1[t * 32 + j];
    hid[j] = a > 0.f ? a : 0.f;
  }
  const float* hs = h + s * Cin;
  for (int c = 0; c < Cin; ++c) {
    float rc = b2[c];
    for (int j = 0; j < 32; ++j) rc += hid[j] * w2[j * Cin + c];
    atomicAdd(msg + d * Cin + c, hs[c] * rc);
  }
}

// ---------------- degree-0 -> degree-1 conv (coord update) ----------------
__global__ void k_conv_vec(const float* x, const float* distC,
                           const int* rs, const int* rd, const int* ls, const int* ld,
                           const float* h,
                           const float* w1, const float* b1,
                           const float* w2, const float* b2,
                           const float* wvec, float* coord) {
  int e = blockIdx.x * blockDim.x + threadIdx.x;
  if (e >= kECROSS) return;
  int s, d, et; cross_edge_idx(e, rs, rd, ls, ld, s, d, et);
  float dist = distC[e];
  float ef[4] = {(et == 0) ? 1.f : 0.f, (et == 1) ? 1.f : 0.f, (et == 2) ? 1.f : 0.f, dist};
  float hid[32];
  for (int j = 0; j < 32; ++j) {
    float a = b1[j];
    for (int t = 0; t < 4; ++t) a += ef[t] * w1[t * 32 + j];
    hid[j] = a > 0.f ? a : 0.f;
  }
  const float* hs = h + s * kEMB;
  float es = 0.f;
  for (int c = 0; c < kEMB; ++c) {
    float rc = b2[c];
    for (int j = 0; j < 32; ++j) rc += hid[j] * w2[j * kEMB + c];
    es += hs[c] * rc * wvec[c];
  }
  float dx = x[d * 3 + 0] - x[s * 3 + 0];
  float dy = x[d * 3 + 1] - x[s * 3 + 1];
  float dz = x[d * 3 + 2] - x[s * 3 + 2];
  float inv = 1.f / (dist + kEPS);
  atomicAdd(coord + d * 3 + 0, es * dx * inv);
  atomicAdd(coord + d * 3 + 1, es * dy * inv);
  atomicAdd(coord + d * 3 + 2, es * dz * inv);
}

// ---------------- output: [1,2,NL,3] = stack(lig_x, x_new[NR:]) ----------------
__global__ void k_out(const float* lig_x, const float* xC, const float* coord, float* out) {
  int i = blockIdx.x * blockDim.x + threadIdx.x;
  if (i >= kNL * 3) return;
  out[i] = lig_x[i];
  out[kNL * 3 + i] = xC[kNR * 3 + i] + coord[kNR * 3 + i];
}

// ---------------- host side ----------------
struct LP {
  const float *b, *g, *rb1, *rb2, *rw1, *rw2, *wk, *wo, *wq, *wv;
  const float *rvb1, *rvb2, *rvw1, *rvw2;
};

extern "C" void kernel_launch(void* const* d_in, const int* in_sizes, int n_in,
                              void* d_out, int out_size, void* d_ws, size_t ws_size,
                              hipStream_t stream) {
  (void)in_sizes; (void)n_in; (void)out_size; (void)ws_size;
  auto F = [&](int i) { return (const float*)d_in[i]; };

  const float* rec_f   = F(0);
  const float* rec_x   = F(1);
  const float* rec_sd  = F(2);
  const float* rec_ew  = F(3);
  const float* lig_f   = F(4);
  const float* lig_x   = F(5);
  const float* lig_ew  = F(6);
  const int P = 7;  // params flattened pytree (dict keys sorted): cross < lig < rec
  // cross.conv (_convvec): r_b1, r_b2, r_w1, r_w2, w_vec
  const float *cvb1 = F(P+0), *cvb2 = F(P+1), *cvw1 = F(P+2), *cvw2 = F(P+3), *cvwv = F(P+4);
  // layer key order: b,g,r_b1,r_b2,r_w1,r_w2,[rv_b1,rv_b2,rv_w1,rv_w2,]wk,wo,wq,wv
  LP crossL[2] = {
    {F(P+5),F(P+6),F(P+7),F(P+8),F(P+9),F(P+10),F(P+11),F(P+12),F(P+13),F(P+14),0,0,0,0},
    {F(P+15),F(P+16),F(P+17),F(P+18),F(P+19),F(P+20),F(P+21),F(P+22),F(P+23),F(P+24),0,0,0,0}};
  // lig.conv (_conv): r_b1,r_b2,r_w1,r_w2,w_edge,w_self
  const float *lcb1 = F(P+25), *lcb2 = F(P+26), *lcw1 = F(P+27), *lcw2 = F(P+28),
              *lcwe = F(P+29), *lcws = F(P+30);
  LP ligL[2] = {
    {F(P+31),F(P+32),F(P+33),F(P+34),F(P+35),F(P+36),F(P+37),F(P+38),F(P+39),F(P+40),0,0,0,0},
    {F(P+41),F(P+42),F(P+43),F(P+44),F(P+45),F(P+46),F(P+47),F(P+48),F(P+49),F(P+50),0,0,0,0}};
  const float *rcb1 = F(P+51), *rcb2 = F(P+52), *rcw1 = F(P+53), *rcw2 = F(P+54),
              *rcwe = F(P+55), *rcws = F(P+56);
  LP recL[2] = {
    {F(P+57),F(P+58),F(P+59),F(P+60),F(P+61),F(P+62),F(P+67),F(P+68),F(P+69),F(P+70),
     F(P+63),F(P+64),F(P+65),F(P+66)},
    {F(P+71),F(P+72),F(P+73),F(P+74),F(P+75),F(P+76),F(P+77),F(P+78),F(P+79),F(P+80),0,0,0,0}};
  const int* rec_src = (const int*)d_in[P + 81];
  const int* rec_dst = (const int*)d_in[P + 82];
  const int* lig_src = (const int*)d_in[P + 83];
  const int* lig_dst = (const int*)d_in[P + 84];

  // ---- workspace carve-up (f32) ----
  float* w = (float*)d_ws;
  float* efeatR = w; w += kER * 5;
  float* vecdotR= w; w += kER;
  float* efeatL = w; w += kEL * 5;
  float* distC  = w; w += kECROSS;
  float* logit  = w; w += kECROSS;
  float* mb     = w; w += kN;
  float* sb     = w; w += kN;
  float* msg    = w; w += kN * kEMB;
  float* qb     = w; w += kN * kEMB;
  float* kb     = w; w += kN * kEMB;
  float* vb     = w; w += kN * kEMB;
  float* h0     = w; w += kN * kEMB;
  float* h1     = w; w += kN * kEMB;
  float* houtR  = w; w += kNR * kEMB;
  float* houtL  = w; w += kNL * kEMB;
  float* xC     = w; w += kN * 3;
  float* coord  = w; w += kN * 3;

  const int TB = 256;
  auto fill = [&](float* p, float v, int n) { k_fill<<<ceil_div(n, TB), TB, 0, stream>>>(p, v, n); };
  auto copy = [&](float* d, const float* s, int n) { k_copy<<<ceil_div(n, TB), TB, 0, stream>>>(d, s, n); };
  const float scale32 = 0.17677669529663687f;  // 1/sqrt(32)
  const float scale64 = 0.125f;                // 1/sqrt(64)

  // =================== per-graph scalar nets ===================
  struct G { const float* f; const float* x; const float* ew; const int* s; const int* d;
             const float* side; float* efeat; float* vecdot; int Nn; int E;
             const LP* L; const float *cb1,*cb2,*cw1,*cw2,*cwe,*cws; float* hout; };
  G gs[2] = {
    {rec_f, rec_x, rec_ew, rec_src, rec_dst, rec_sd, efeatR, vecdotR, kNR, kER,
     recL, rcb1, rcb2, rcw1, rcw2, rcwe, rcws, houtR},
    {lig_f, lig_x, lig_ew, lig_src, lig_dst, nullptr, efeatL, nullptr, kNL, kEL,
     ligL, lcb1, lcb2, lcw1, lcw2, lcwe, lcws, houtL}};

  for (int gi = 0; gi < 2; ++gi) {
    const G& g = gs[gi];
    const int gE = ceil_div(g.E, TB);
    k_geom<<<gE, TB, 0, stream>>>(g.x, g.ew, g.s, g.d, g.side, g.efeat, g.vecdot, g.E);
    copy(h0, g.f, g.Nn * kC);
    for (int li = 0; li < 2; ++li) {
      const LP& L = g.L[li];
      gemm(h0, L.wq, nullptr, qb, g.Nn, kC, kC, stream);
      gemm(h0, L.wk, nullptr, kb, g.Nn, kC, kC, stream);
      gemm(h0, L.wv, nullptr, vb, g.Nn, kC, kC, stream);
      fill(mb, -3.0e38f, g.Nn); fill(sb, 0.f, g.Nn); fill(msg, 0.f, g.Nn * kC);
      k_attn_pass1<false><<<gE, TB, 0, stream>>>(g.efeat, g.s, g.d, nullptr, nullptr,
          qb, kb, L.rw1, L.rb1, L.rw2, L.rb2, logit, mb, g.E, kC, scale32);
      k_attn_pass2<false><<<gE, TB, 0, stream>>>(g.s, g.d, nullptr, nullptr, logit, mb, sb, g.E);
      const float* vd = (li == 0) ? g.vecdot : nullptr;
      k_attn_pass3<false><<<gE, TB, 0, stream>>>(g.efeat, g.s, g.d, nullptr, nullptr,
          vb, L.rw1, L.rb1, L.rw2, L.rb2, vd, L.rvw1, L.rvb1, L.rvw2, L.rvb2,
          logit, sb, msg, g.E, kC);
      gemm(msg, L.wo, h0, h1, g.Nn, kC, kC, stream);                 // h + msg@wo
      k_gnorm<<<ceil_div(g.Nn * kC, TB), TB, 0, stream>>>(h1, L.g, L.b, h0, g.Nn * kC, kC);
    }
    // GConvSE3: h@w_self + segsum(h[src]*r)@w_edge
    fill(msg, 0.f, g.Nn * kC);
    k_conv_msg<<<gE, TB, 0, stream>>>(g.efeat, g.s, g.d, h0, g.cw1, g.cb1, g.cw2, g.cb2,
                                      msg, g.E, kC);
    gemm(h0, g.cws, nullptr, h1, g.Nn, kEMB, kC, stream);
    gemm(msg, g.cwe, h1, g.hout, g.Nn, kEMB, kC, stream);
  }

  // =================== cross net on combined graph ===================
  copy(xC, rec_x, kNR * 3);
  copy(xC + kNR * 3, lig_x, kNL * 3);
  copy(h0, houtR, kNR * kEMB);
  copy(h0 + kNR * kEMB, houtL, kNL * kEMB);
  const int gEc = ceil_div(kECROSS, TB);
  k_cross_geom<<<gEc, TB, 0, stream>>>(xC, rec_src, rec_dst, lig_src, lig_dst, distC);

  for (int li = 0; li < 2; ++li) {
    const LP& L = crossL[li];
    gemm(h0, L.wq, nullptr, qb, kN, kEMB, kEMB, stream);
    gemm(h0, L.wk, nullptr, kb, kN, kEMB, kEMB, stream);
    gemm(h0, L.wv, nullptr, vb, kN, kEMB, kEMB, stream);
    fill(mb, -3.0e38f, kN); fill(sb, 0.f, kN); fill(msg, 0.f, kN * kEMB);
    k_attn_pass1<true><<<gEc, TB, 0, stream>>>(distC, rec_src, rec_dst, lig_src, lig_dst,
        qb, kb, L.rw1, L.rb1, L.rw2, L.rb2, logit, mb, kECROSS, kEMB, scale64);
    k_attn_pass2<true><<<gEc, TB, 0, stream>>>(rec_src, rec_dst, lig_src, lig_dst,
        logit, mb, sb, kECROSS);
    k_attn_pass3<true><<<gEc, TB, 0, stream>>>(distC, rec_src, rec_dst, lig_src, lig_dst,
        vb, L.rw1, L.rb1, L.rw2, L.rb2, nullptr, nullptr, nullptr, nullptr, nullptr,
        logit, sb, msg, kECROSS, kEMB);
    gemm(msg, L.wo, h0, h1, kN, kEMB, kEMB, stream);
    k_gnorm<<<ceil_div(kN * kEMB, TB), TB, 0, stream>>>(h1, L.g, L.b, h0, kN * kEMB, kEMB);
  }

  fill(coord, 0.f, kN * 3);
  k_conv_vec<<<gEc, TB, 0, stream>>>(xC, distC, rec_src, rec_dst, lig_src, lig_dst,
                                     h0, cvw1, cvb1, cvw2, cvb2, cvwv, coord);
  k_out<<<ceil_div(kNL * 3, TB), TB, 0, stream>>>(lig_x, xC, coord, (float*)d_out);
}
